// ModulatedConv1d_72524817760949
// MI455X (gfx1250) — matrix operand order repaired and checked by reading, not compiled, not run on hardware
//
#include <hip/hip_runtime.h>

#define IN_C   512
#define OUT_C  512
#define KW     3
#define LLEN   2048
#define BATCH  16
#define STYLE  512
#define LPAD   (LLEN + 2)     // padded l rows in Xt: row 0 and row LLEN+1 are zeros

#define SCALE_F     0.014731391274719739f   // 1/sqrt(512*9)
#define MOD_SCALE_F 0.04419417382415922f    // 1/sqrt(512)
#define EPS_F       1e-8f

typedef __bf16 bf16_t;
typedef bf16_t v16bf __attribute__((ext_vector_type(16)));
typedef bf16_t v8bf  __attribute__((ext_vector_type(8)));
typedef float  v8f   __attribute__((ext_vector_type(8)));
typedef int    v4i   __attribute__((vector_size(16)));

__device__ __forceinline__ bf16_t f2bf(float f) {
  unsigned u = __builtin_bit_cast(unsigned, f);
  unsigned r = u + 0x7FFFu + ((u >> 16) & 1u);   // round-to-nearest-even
  unsigned short h = (unsigned short)(r >> 16);
  return __builtin_bit_cast(bf16_t, h);
}

__device__ __forceinline__ bf16_t bf_zero() {
  unsigned short h = 0;
  return __builtin_bit_cast(bf16_t, h);
}

// ---- CDNA5 async global->LDS copy (ASYNCcnt-tracked), with fallback --------
#if __has_builtin(__builtin_amdgcn_global_load_async_to_lds_b128)
__device__ __forceinline__ void async_b128(const bf16_t* g, bf16_t* l) {
  __builtin_amdgcn_global_load_async_to_lds_b128(
      (__attribute__((address_space(1))) v4i*)g,
      (__attribute__((address_space(3))) v4i*)l,
      /*offset=*/0, /*cpol=*/0);
}
#else
__device__ __forceinline__ void async_b128(const bf16_t* g, bf16_t* l) {
  *(v8bf*)l = *(const v8bf*)g;
}
#endif

__device__ __forceinline__ void wait_async_all() {
#if __has_builtin(__builtin_amdgcn_s_wait_asynccnt)
  __builtin_amdgcn_s_wait_asynccnt(0);
#else
  asm volatile("s_wait_asynccnt 0x0" ::: "memory");
#endif
}

// ---------------------------------------------------------------------------
// Kernel 1: s[b][ic] = (style[b] . mod_weight[ic]) * MOD_SCALE + mod_bias[ic]
// ---------------------------------------------------------------------------
__global__ __launch_bounds__(256)
void mod_s_kernel(const float* __restrict__ style,
                  const float* __restrict__ mw,
                  const float* __restrict__ mb,
                  float* __restrict__ s_out) {
  const int b = blockIdx.x;
  __shared__ float st[STYLE];
  for (int j = threadIdx.x; j < STYLE; j += blockDim.x)
    st[j] = style[b * STYLE + j];
  __syncthreads();
  for (int ic = threadIdx.x; ic < IN_C; ic += blockDim.x) {
    const float* w = mw + (size_t)ic * STYLE;
    float acc = 0.0f;
    for (int j = 0; j < STYLE; ++j) acc += st[j] * w[j];
    s_out[b * IN_C + ic] = acc * MOD_SCALE_F + mb[ic];
  }
}

// ---------------------------------------------------------------------------
// Kernel 2: one wave per (b, oc). Modulate, demodulate, emit bf16 weights
// laid out [b][tap][oc][ic] (ic contiguous -> WMMA A-fragment friendly).
// ---------------------------------------------------------------------------
__global__ __launch_bounds__(256)
void mod_w_kernel(const float* __restrict__ weight,
                  const float* __restrict__ s,
                  bf16_t* __restrict__ wout) {
  const int wid  = (int)((blockIdx.x * blockDim.x + threadIdx.x) >> 5);
  const int lane = threadIdx.x & 31;
  const int b  = wid >> 9;      // / 512
  const int oc = wid & 511;
  if (b >= BATCH) return;

  const float* wrow = weight + (size_t)oc * (IN_C * KW);
  const float* sb   = s + (size_t)b * IN_C;

  float wv[48];
  float sum = 0.0f;
#pragma unroll
  for (int i = 0; i < 48; ++i) {
    int e  = lane + i * 32;          // e in [0, 1536)
    int ic = e / 3;
    float v = SCALE_F * wrow[e] * sb[ic];
    wv[i] = v;
    sum += v * v;
  }
#pragma unroll
  for (int off = 16; off > 0; off >>= 1)
    sum += __shfl_xor(sum, off, 32);
  const float demod = rsqrtf(sum + EPS_F);

#pragma unroll
  for (int i = 0; i < 48; ++i) {
    int e  = lane + i * 32;
    int ic = e / 3;
    int k  = e - ic * 3;
    wout[(((size_t)b * KW + k) * OUT_C + oc) * IN_C + ic] = f2bf(wv[i] * demod);
  }
}

// ---------------------------------------------------------------------------
// Kernel 3: transpose + convert input:  x[b][ic][l] f32 -> Xt[b][1+l][ic] bf16
// Row 0 and row LLEN+1 of each batch are zero (conv halo), so the GEMM needs
// no bounds checks and can stage X purely with async b128 copies.
// ---------------------------------------------------------------------------
__global__ __launch_bounds__(256)
void xpose_kernel(const float* __restrict__ x, bf16_t* __restrict__ xt) {
  const int b   = blockIdx.z;
  const int ic0 = blockIdx.y * 64;
  const int l0  = blockIdx.x * 64;
  const int tid = threadIdx.x;
  __shared__ __align__(16) bf16_t T[64 * 72];

  // read 64(ic) x 64(l) tile, coalesced along l, f32->bf16, transpose into LDS
  for (int c = tid; c < 64 * 16; c += 256) {      // 1024 float4 chunks
    int ici = c >> 4;
    int li4 = (c & 15) << 2;
    const float4 v = *(const float4*)&x[((size_t)b * IN_C + ic0 + ici) * LLEN + l0 + li4];
    T[(li4 + 0) * 72 + ici] = f2bf(v.x);
    T[(li4 + 1) * 72 + ici] = f2bf(v.y);
    T[(li4 + 2) * 72 + ici] = f2bf(v.z);
    T[(li4 + 3) * 72 + ici] = f2bf(v.w);
  }
  __syncthreads();

  // write [l][ic] tile, coalesced along ic, 16B chunks
  for (int c = tid; c < 64 * 8; c += 256) {       // 512 v8bf chunks
    int li   = c >> 3;
    int coff = (c & 7) << 3;
    v8bf v = *(const v8bf*)&T[li * 72 + coff];
    *(v8bf*)&xt[((size_t)b * LPAD + 1 + l0 + li) * IN_C + ic0 + coff] = v;
  }

  // zero halo rows once per batch edge
  v8bf z;
#pragma unroll
  for (int e = 0; e < 8; ++e) z[e] = bf_zero();
  if (l0 == 0 && tid < 8)
    *(v8bf*)&xt[((size_t)b * LPAD + 0) * IN_C + ic0 + (tid << 3)] = z;
  if (l0 == LLEN - 64 && tid < 8)
    *(v8bf*)&xt[((size_t)b * LPAD + LLEN + 1) * IN_C + ic0 + (tid << 3)] = z;
}

// ---------------------------------------------------------------------------
// Kernel 4: per-batch GEMM with 3-tap accumulation via WMMA bf16.
//   O[oc, l] = sum_t sum_ic W[b][t][oc][ic] * Xt[b][1 + l + t - 1][ic]
// Block: 128(oc) x 128(l), 256 threads (8 waves), wave = 64x32 (4x2 tiles).
// Double-buffered LDS; staging via async global->LDS b128 copies.
// ---------------------------------------------------------------------------
#define BM 128
#define BN 128
#define KC 64
#define KCP 72          // padded LDS row stride (16B-aligned rows, low conflict)
#define NTHREADS 256
#define NSTEPS (IN_C / KC)

__device__ __forceinline__ void stage_async(const bf16_t* __restrict__ wsrc,
                                            const bf16_t* __restrict__ xsrc,
                                            bf16_t* wdst, bf16_t* xdst, int tid) {
  // W: 3 taps x 128 oc x 64 ic = 3072 b128 chunks
  for (int c = tid; c < KW * BM * (KC / 8); c += NTHREADS) {
    int tap = c >> 10;
    int rem = c & 1023;
    int row = rem >> 3;
    int col = (rem & 7) << 3;
    async_b128(wsrc + ((size_t)tap * OUT_C + row) * IN_C + col,
               wdst + (tap * BM + row) * KCP + col);
  }
  // X: 130 l-rows x 64 ic = 1040 b128 chunks (halo rows are pre-zeroed in Xt)
  for (int c = tid; c < (BN + 2) * (KC / 8); c += NTHREADS) {
    int row = c >> 3;
    int col = (c & 7) << 3;
    async_b128(xsrc + (size_t)row * IN_C + col,
               xdst + row * KCP + col);
  }
}

__global__ __launch_bounds__(NTHREADS)
void conv_wmma_kernel(const bf16_t* __restrict__ xt,
                      const bf16_t* __restrict__ wmod,
                      float* __restrict__ out) {
  __shared__ __align__(16) bf16_t Ws[2][KW * BM * KCP];   // 2 x 55,296 B
  __shared__ __align__(16) bf16_t Xs[2][(BN + 2) * KCP];  // 2 x 18,720 B

  const int tid = threadIdx.x;
  const int b   = blockIdx.z;
  const int oc0 = blockIdx.y * BM;
  const int l0  = blockIdx.x * BN;

  const int lane = tid & 31;
  const int wave = tid >> 5;
  const int wm   = wave & 1;    // oc direction: 2 waves
  const int wn   = wave >> 1;   // l  direction: 4 waves
  const int r    = lane & 15;
  const int hi   = lane >> 4;

  // global bases for this block (ic advances by KC each step)
  const bf16_t* wbase = wmod + (((size_t)b * KW) * OUT_C + oc0) * IN_C;
  const bf16_t* xbase = xt + ((size_t)b * LPAD + l0) * IN_C;   // row li -> l0-1+li (+1 pad)

  v8f acc[4][2];
#pragma unroll
  for (int i = 0; i < 4; ++i)
#pragma unroll
    for (int j = 0; j < 2; ++j) {
      v8f z = {0.f, 0.f, 0.f, 0.f, 0.f, 0.f, 0.f, 0.f};
      acc[i][j] = z;
    }

  // prologue: stage chunk 0 into buffer 0
  stage_async(wbase, xbase, &Ws[0][0], &Xs[0][0], tid);
  wait_async_all();
  __syncthreads();

  for (int step = 0; step < NSTEPS; ++step) {
    const int cur = step & 1;

    // overlap: issue async stage of next K-chunk into the other buffer
    if (step + 1 < NSTEPS) {
      const int ic0 = (step + 1) * KC;
      stage_async(wbase + ic0, xbase + ic0, &Ws[1 - cur][0], &Xs[1 - cur][0], tid);
    }

    const bf16_t* Wb = &Ws[cur][0];
    const bf16_t* Xb = &Xs[cur][0];

#pragma unroll
    for (int kk = 0; kk < KC; kk += 32) {
#pragma unroll
      for (int t = 0; t < KW; ++t) {
        // B fragments: lane = column l, 16 contiguous ic per lane
        v16bf bfr[2];
#pragma unroll
        for (int j = 0; j < 2; ++j) {
          const bf16_t* p = &Xb[(wn * 32 + j * 16 + r + t) * KCP + kk + hi * 16];
          v8bf c0 = *(const v8bf*)p;
          v8bf c1 = *(const v8bf*)(p + 8);
          bfr[j] = __builtin_shufflevector(c0, c1,
                     0,1,2,3,4,5,6,7,8,9,10,11,12,13,14,15);
        }
#pragma unroll
        for (int i = 0; i < 4; ++i) {
          // A fragment: lane = row oc, 16 contiguous K at hi*16
          const bf16_t* p = &Wb[(t * BM + wm * 64 + i * 16 + r) * KCP + kk + hi * 16];
          v8bf a0 = *(const v8bf*)p;
          v8bf a1 = *(const v8bf*)(p + 8);
          v16bf af = __builtin_shufflevector(a0, a1,
                       0,1,2,3,4,5,6,7,8,9,10,11,12,13,14,15);
#pragma unroll
          for (int j = 0; j < 2; ++j)
            acc[i][j] = __builtin_amdgcn_wmma_f32_16x16x32_bf16(
                false, af, false, bfr[j], (short)0, acc[i][j], false, false);
        }
      }
    }

    wait_async_all();   // next buffer's async copies done before the barrier
    __syncthreads();
  }

  // ---- epilogue: C layout — lane&15 = N, hi*8 + vgpr = M
  const int n  = r;
  const int mo = hi * 8;
#pragma unroll
  for (int i = 0; i < 4; ++i)
#pragma unroll
    for (int j = 0; j < 2; ++j) {
      const int oc = oc0 + wm * 64 + i * 16 + mo;
      const int l  = l0 + wn * 32 + j * 16 + n;
#pragma unroll
      for (int rr = 0; rr < 8; ++rr)
        out[((size_t)b * OUT_C + (oc + rr)) * LLEN + l] = acc[i][j][rr];
    }
}

// ---------------------------------------------------------------------------
extern "C" void kernel_launch(void* const* d_in, const int* in_sizes, int n_in,
                              void* d_out, int out_size, void* d_ws, size_t ws_size,
                              hipStream_t stream) {
  (void)in_sizes; (void)n_in; (void)out_size; (void)ws_size;
  const float* x      = (const float*)d_in[0];
  const float* style  = (const float*)d_in[1];
  const float* weight = (const float*)d_in[2];
  const float* mw     = (const float*)d_in[3];
  const float* mb     = (const float*)d_in[4];
  float* out = (float*)d_out;

  // workspace layout
  float*  s_ws = (float*)d_ws;                                   // 32 KB
  bf16_t* w_ws = (bf16_t*)((char*)d_ws + (64u << 10));           // 24 MB
  bf16_t* xt_ws = (bf16_t*)((char*)d_ws + (64u << 10)
                            + (size_t)BATCH * KW * OUT_C * IN_C * sizeof(bf16_t)); // 33.6 MB

  mod_s_kernel<<<dim3(BATCH), dim3(256), 0, stream>>>(style, mw, mb, s_ws);
  mod_w_kernel<<<dim3((BATCH * OUT_C * 32) / 256), dim3(256), 0, stream>>>(weight, s_ws, w_ws);
  xpose_kernel<<<dim3(LLEN / 64, IN_C / 64, BATCH), dim3(256), 0, stream>>>(x, xt_ws);
  conv_wmma_kernel<<<dim3(LLEN / BN, OUT_C / BM, BATCH), dim3(256), 0, stream>>>(xt_ws, w_ws, out);
}
